// LSTM_8555574854051
// MI455X (gfx1250) — matrix operand order repaired and checked by reading, not compiled, not run on hardware
//
#include <hip/hip_runtime.h>
#include <hip/hip_bf16.h>

// Problem constants (match reference).
#define B_ 256
#define T_ 128
#define H_ 1024
#define C_ 10

typedef __attribute__((ext_vector_type(16))) __bf16 v16bf;
typedef __attribute__((ext_vector_type(8)))  __bf16 v8bf;
typedef __attribute__((ext_vector_type(8)))  float  v8f;

// ---------- fast transcendentals (native v_exp_f32 / v_rcp_f32) ----------
__device__ __forceinline__ float fsigmoid(float x) {
  // 1 / (1 + 2^(-x*log2e))
  return __builtin_amdgcn_rcpf(1.0f + __builtin_amdgcn_exp2f(-1.4426950408889634f * x));
}
__device__ __forceinline__ float ftanh(float x) {
  // tanh(x) = 1 - 2/(1 + 2^(2x*log2e))
  return 1.0f - 2.0f * __builtin_amdgcn_rcpf(1.0f + __builtin_amdgcn_exp2f(2.8853900817779268f * x));
}

// ---------- pack recurrent weights into WMMA B-matrix layout (bf16) ----------
// Dest element index = gate*H*H + (nb*32 + kb)*512 + lane*16 + e
//   lane<16 : B[k0+(e&7)+(e&8?16:0)][n0+lane]
//   lane>=16: B[k0+8+(e&7)+(e&8?16:0)][n0+lane-16]
__global__ void lstm_pack_weights(const float* __restrict__ Wg,
                                  const float* __restrict__ Wi,
                                  const float* __restrict__ Wf,
                                  const float* __restrict__ Wo,
                                  __bf16* __restrict__ wpack) {
  int idx = blockIdx.x * blockDim.x + threadIdx.x;
  const int HH = H_ * H_;
  if (idx >= 4 * HH) return;
  int gate = idx / HH;
  int rem  = idx % HH;
  int blk  = rem >> 9;       // / 512
  int win  = rem & 511;
  int lane = win >> 4;
  int e    = win & 15;
  int nb   = blk >> 5;       // / (H/32)
  int kb   = blk & 31;
  int n = nb * 16 + (lane & 15);
  int k = kb * 32 + ((lane < 16) ? 0 : 8) + (e & 7) + ((e & 8) ? 16 : 0);
  const float* W = (gate == 0) ? Wg : (gate == 1) ? Wi : (gate == 2) ? Wf : Wo;
  wpack[idx] = (__bf16)W[k * H_ + n];
}

// ---------- zero-init h (both buffers) and c ----------
__global__ void lstm_init_state(__bf16* __restrict__ h0,
                                __bf16* __restrict__ h1,
                                float* __restrict__ c) {
  int idx = blockIdx.x * blockDim.x + threadIdx.x;
  if (idx < B_ * H_) {
    h0[idx] = (__bf16)0.0f;
    h1[idx] = (__bf16)0.0f;
    c[idx]  = 0.0f;
  }
}

// ---------- one LSTM timestep, fully fused ----------
// grid = (B/64, H/16), block = 128 (4 waves). Wave w owns m-tile blockIdx.x*4+w,
// all 4 waves share n-tile blockIdx.y (weight lines fetched once per WGP).
__global__ void __launch_bounds__(128)
lstm_step_kernel(const float* __restrict__ x,
                 const __bf16* __restrict__ wpack,
                 const __bf16* __restrict__ h_prev,
                 __bf16* __restrict__ h_next,
                 float* __restrict__ c_state,
                 const float* __restrict__ Wgx, const float* __restrict__ Wix,
                 const float* __restrict__ Wfx, const float* __restrict__ Wox,
                 const float* __restrict__ bg,  const float* __restrict__ bi,
                 const float* __restrict__ bf_, const float* __restrict__ bo,
                 int t) {
  const int lane = threadIdx.x & 31;
  const int wave = threadIdx.x >> 5;
  const int m0 = (blockIdx.x * 4 + wave) * 16;
  const int nb = blockIdx.y;
  const int n0 = nb * 16;
  const int HH = H_ * H_;

  v8f ag = {}, ai = {}, af = {}, ao = {};

  // A-matrix 16x32 bf16 layout: lane<16 rows M=lane, K {0..7, 16..23};
  // lane>=16 rows M=lane-16, K {8..15, 24..31}.
  const int base8 = (lane & 16) ? 8 : 0;
  const __bf16* arow = h_prev + (size_t)(m0 + (lane & 15)) * H_ + base8;

  const __bf16* pg = wpack + (size_t)nb * 32 * 512 + lane * 16;
  const __bf16* pi = pg + HH;
  const __bf16* pf = pi + HH;
  const __bf16* po = pf + HH;

#pragma unroll 2
  for (int kb = 0; kb < H_ / 32; ++kb) {
    v8bf alo = *(const v8bf*)(arow);
    v8bf ahi = *(const v8bf*)(arow + 16);
    arow += 32;
    v16bf a = __builtin_shufflevector(alo, ahi,
        0, 1, 2, 3, 4, 5, 6, 7, 8, 9, 10, 11, 12, 13, 14, 15);
    v16bf bgt = *(const v16bf*)pg; pg += 512;
    v16bf bit = *(const v16bf*)pi; pi += 512;
    v16bf bft = *(const v16bf*)pf; pf += 512;
    v16bf bot = *(const v16bf*)po; po += 512;
    ag = __builtin_amdgcn_wmma_f32_16x16x32_bf16(false, a, false, bgt, (short)0, ag, false, false);
    ai = __builtin_amdgcn_wmma_f32_16x16x32_bf16(false, a, false, bit, (short)0, ai, false, false);
    af = __builtin_amdgcn_wmma_f32_16x16x32_bf16(false, a, false, bft, (short)0, af, false, false);
    ao = __builtin_amdgcn_wmma_f32_16x16x32_bf16(false, a, false, bot, (short)0, ao, false, false);
  }

  // Epilogue: z = acc + x_t * Wx + b; gates; c/h update. D layout:
  // lane<16: VGPR r -> (M=r, N=lane); lane>=16: (M=8+r, N=lane-16).
  const int n = n0 + (lane & 15);
  const float vbg = bg[n],  vbi = bi[n],  vbf = bf_[n], vbo = bo[n];
  const float wgx = Wgx[n], wix = Wix[n], wfx = Wfx[n], wox = Wox[n];
  const int mo = m0 + ((lane & 16) ? 8 : 0);

#pragma unroll
  for (int r = 0; r < 8; ++r) {
    const int m = mo + r;
    const float xv = x[m * T_ + t];
    const float zg = ag[r] + xv * wgx + vbg;
    const float zi = ai[r] + xv * wix + vbi;
    const float zf = af[r] + xv * wfx + vbf;
    const float zo = ao[r] + xv * wox + vbo;
    const float gv = ftanh(zg);
    const float iv = fsigmoid(zi);
    const float fv = fsigmoid(zf);
    const float ov = fsigmoid(zo);
    const size_t ci = (size_t)m * H_ + n;
    const float cn = gv * iv + c_state[ci] * fv;
    c_state[ci] = cn;
    h_next[ci] = (__bf16)(ftanh(cn) * ov);
  }
}

// ---------- final projection: out[B,C] = h @ W_ph + b_p (tiny) ----------
__global__ void lstm_proj_kernel(const __bf16* __restrict__ h,
                                 const float* __restrict__ Wp,
                                 const float* __restrict__ bp,
                                 float* __restrict__ out) {
  int id = blockIdx.x * blockDim.x + threadIdx.x;
  if (id >= B_ * C_) return;
  int b = id / C_, c = id % C_;
  const __bf16* hr = h + (size_t)b * H_;
  float acc = 0.0f;
#pragma unroll 8
  for (int k = 0; k < H_; ++k) acc += (float)hr[k] * Wp[k * C_ + c];
  out[id] = acc + bp[c];
}

extern "C" void kernel_launch(void* const* d_in, const int* in_sizes, int n_in,
                              void* d_out, int out_size, void* d_ws, size_t ws_size,
                              hipStream_t stream) {
  (void)in_sizes; (void)n_in; (void)out_size; (void)ws_size;
  const float* x   = (const float*)d_in[0];
  const float* Wgx = (const float*)d_in[1];
  const float* Wix = (const float*)d_in[2];
  const float* Wfx = (const float*)d_in[3];
  const float* Wox = (const float*)d_in[4];
  const float* Wgh = (const float*)d_in[5];
  const float* Wih = (const float*)d_in[6];
  const float* Wfh = (const float*)d_in[7];
  const float* Woh = (const float*)d_in[8];
  const float* Wph = (const float*)d_in[9];
  const float* bg  = (const float*)d_in[10];
  const float* bi  = (const float*)d_in[11];
  const float* bf_ = (const float*)d_in[12];
  const float* bo  = (const float*)d_in[13];
  const float* bp  = (const float*)d_in[14];
  float* out = (float*)d_out;

  // Workspace layout (bytes):
  //   [0, 8MB)        packed bf16 weights (4 gates)
  //   [8MB, 8.5MB)    h ping (bf16)
  //   [8.5MB, 9MB)    h pong (bf16)
  //   [9MB, 10MB)     c state (f32)
  char* ws = (char*)d_ws;
  __bf16* wpack = (__bf16*)(ws);
  __bf16* h0    = (__bf16*)(ws + (size_t)4 * H_ * H_ * 2);
  __bf16* h1    = (__bf16*)(ws + (size_t)4 * H_ * H_ * 2 + (size_t)B_ * H_ * 2);
  float*  c     = (float*)(ws + (size_t)4 * H_ * H_ * 2 + (size_t)2 * B_ * H_ * 2);

  // 1) Pack weights into WMMA B layout (once per call; deterministic).
  {
    int total = 4 * H_ * H_;
    lstm_pack_weights<<<(total + 255) / 256, 256, 0, stream>>>(Wgh, Wih, Wfh, Woh, wpack);
  }
  // 2) Zero h/c.
  {
    int total = B_ * H_;
    lstm_init_state<<<(total + 255) / 256, 256, 0, stream>>>(h0, h1, c);
  }
  // 3) Recurrence: one fused GEMM+gate kernel per timestep (graph-captured).
  dim3 grid(B_ / 64, H_ / 16);
  for (int t = 0; t < T_; ++t) {
    const __bf16* hp = (t & 1) ? h1 : h0;
    __bf16*       hn = (t & 1) ? h0 : h1;
    lstm_step_kernel<<<grid, 128, 0, stream>>>(x, wpack, hp, hn, c,
                                               Wgx, Wix, Wfx, Wox,
                                               bg, bi, bf_, bo, t);
  }
  // After T=128 (even) steps the latest h lives in h0.
  const __bf16* hfinal = (T_ % 2 == 0) ? h0 : h1;
  // 4) Projection.
  lstm_proj_kernel<<<(B_ * C_ + 255) / 256, 256, 0, stream>>>(hfinal, Wph, bp, out);
}
